// VisionRetentionChunk_34497177322043
// MI455X (gfx1250) — matrix operand order repaired and checked by reading, not compiled, not run on hardware
//
#include <hip/hip_runtime.h>

typedef __attribute__((ext_vector_type(16))) _Float16 v16h;
typedef __attribute__((ext_vector_type(8)))  _Float16 v8h;
typedef __attribute__((ext_vector_type(8)))  float    v8f;

#define C_DIM   512
#define NH      8
#define KD      64
#define SCALING 0.125f   // KD^-0.5

static __device__ __forceinline__ v8f wmma_f16(v16h a, v16h b, v8f c) {
  // D = A(16x32 f16) * B(32x16 f16) + C(16x16 f32)
  return __builtin_amdgcn_wmma_f32_16x16x32_f16(false, a, false, b, (short)0, c,
                                                false, false);
}

// ---------------------------------------------------------------- elementwise
__global__ __launch_bounds__(256) void cvt_f32_f16(const float* __restrict__ in,
                                                   _Float16* __restrict__ out,
                                                   size_t n) {
  size_t i = (size_t)blockIdx.x * 256 + threadIdx.x;
  if (i < n) out[i] = (_Float16)in[i];
}

// weight convert + transpose: out[n*K+k] = (f16) in[k*N+n]   (512x512)
__global__ __launch_bounds__(256) void cvt_transpose_f16(
    const float* __restrict__ in, _Float16* __restrict__ out) {
  int i = blockIdx.x * 256 + threadIdx.x;   // output-linear: coalesced stores
  int n = i >> 9, k = i & (C_DIM - 1);
  out[i] = (_Float16)in[(size_t)k * C_DIM + n];
}

// 5x5 depthwise conv, pad 2, NHWC
__global__ __launch_bounds__(256) void dwconv5x5(
    const float* __restrict__ V, const float* __restrict__ Wt,
    const float* __restrict__ Bi, float* __restrict__ Out, size_t n) {
  size_t i = (size_t)blockIdx.x * 256 + threadIdx.x;
  if (i >= n) return;
  int    c  = (int)(i & (C_DIM - 1));
  size_t sp = i >> 9;
  int    w  = (int)(sp & 63);
  int    h  = (int)((sp >> 6) & 63);
  size_t b  = sp >> 12;
  float acc = Bi[c];
  #pragma unroll
  for (int dy = 0; dy < 5; ++dy) {
    int hy = h + dy - 2;
    if (hy < 0 || hy > 63) continue;
    #pragma unroll
    for (int dx = 0; dx < 5; ++dx) {
      int wx = w + dx - 2;
      if (wx < 0 || wx > 63) continue;
      acc += V[(((b * 64 + hy) * 64) + wx) * C_DIM + c] *
             Wt[(dy * 5 + dx) * C_DIM + c];
    }
  }
  Out[i] = acc;
}

// ---------------------------------------------------------------- WMMA GEMM
// acc = A[M,K]_f16 @ Bt[N,K]_f16^T ; val = (acc + bias[n]) * scale
// mode 0: Out32 = val
// mode 1: rope epilogue -> Out16 = theta_shift(val) (f16)
// mode 2: Out32 = val and Out16 = (f16)val
// block = 256 threads (8 waves), block tile = 128(M) x 64(N); each wave owns a
// 16x64 strip -> 4 WMMAs per A fragment. B staged as 64x256-half slabs in LDS
// (33.8KB, 264-half pitch => conflict-free b128 fragment reads).
__global__ __launch_bounds__(256) void gemm_wmma_f16(
    const _Float16* __restrict__ A, const _Float16* __restrict__ Bt,
    const float* __restrict__ bias, float* __restrict__ Out32,
    _Float16* __restrict__ Out16, const float* __restrict__ sn,
    const float* __restrict__ cs, int N, int K, float scale, int mode) {
  __shared__ _Float16 BsT[64][264];   // [n][k-slab], padded pitch 528B
  const int    lane = threadIdx.x & 31;
  const int    wave = threadIdx.x >> 5;
  const int    hb   = lane >> 4;
  const int    l15  = lane & 15;
  const int    n0   = blockIdx.x * 64;
  const size_t m0   = (size_t)blockIdx.y * 128 + (size_t)wave * 16;

  v8f acc[4] = {};
  const _Float16* ap0 = A + (m0 + l15) * (size_t)K + 8 * hb;

  for (int kb = 0; kb < K; kb += 256) {
    __syncthreads();
    {  // stage B slab: 64 rows x 256 halfs, 64 halfs per thread
      int nn = threadIdx.x >> 2;          // 0..63
      int kk = (threadIdx.x & 3) * 64;    // 0..192
      const _Float16* src = Bt + (size_t)(n0 + nn) * K + kb + kk;
      #pragma unroll
      for (int u = 0; u < 8; ++u)
        *(v8h*)&BsT[nn][kk + u * 8] = *(const v8h*)(src + u * 8);
    }
    __syncthreads();

    #pragma unroll 2
    for (int k0 = 0; k0 < 256; k0 += 32) {
      union { v16h v; v8h h[2]; } a;
      a.h[0] = *(const v8h*)(ap0 + kb + k0);        // K = +8hb+{0..7}
      a.h[1] = *(const v8h*)(ap0 + kb + k0 + 16);   // K = +16+8hb+{0..7}
      __builtin_prefetch(ap0 + kb + k0 + 32, 0, 1);
      #pragma unroll
      for (int nt = 0; nt < 4; ++nt) {
        union { v16h v; v8h h[2]; } b;
        const _Float16* bp = &BsT[nt * 16 + l15][k0 + 8 * hb];
        b.h[0] = *(const v8h*)bp;
        b.h[1] = *(const v8h*)(bp + 16);
        acc[nt] = wmma_f16(a.v, b.v, acc[nt]);
      }
    }
  }

  #pragma unroll
  for (int nt = 0; nt < 4; ++nt) {
    const int   col  = n0 + nt * 16 + l15;
    const float bcol = bias[col];
    #pragma unroll
    for (int r = 0; r < 8; ++r) {
      size_t row = m0 + r + 8 * hb;            // token index
      float  val = (acc[nt][r] + bcol) * scale;
      if (mode == 1) {
        // RoPE: partner channel lives in adjacent lane of the fragment
        float partner = __shfl_xor(val, 1);
        int   w  = (int)(row & 63);
        int   h  = (int)((row >> 6) & 63);
        int   dk = col & (KD - 1);
        float ang_s = sn[((size_t)(h * 64 + w)) * KD + dk];
        float ang_c = cs[((size_t)(h * 64 + w)) * KD + dk];
        float o = (col & 1) ? (val * ang_c + partner * ang_s)
                            : (val * ang_c - partner * ang_s);
        Out16[row * N + col] = (_Float16)o;
      } else {
        Out32[row * N + col] = val;
        if (mode == 2) Out16[row * N + col] = (_Float16)val;
      }
    }
  }
}

// ---------------------------------------------------------------- attention
// One block per (b, fixed-position, head). 64x64x64 tile fully in LDS:
// S = Q K^T (WMMA), softmax(+mask) with 4 threads/row, O = P V (WMMA).
// axis 0: sequence over W (stride C); axis 1: over H (stride W*C).
// Output f16; optional f32 addend (lepe) fused into the epilogue.
__global__ __launch_bounds__(256) void attn_axis(
    const _Float16* __restrict__ Qg, const _Float16* __restrict__ Kg,
    const _Float16* __restrict__ Vg, const float* __restrict__ mask,
    _Float16* __restrict__ out16, const float* __restrict__ addend, int axis) {
  __shared__ _Float16 Qs[64][72];
  __shared__ _Float16 Ks[64][72];
  __shared__ _Float16 VsT[64][72];    // transposed: VsT[d][v]
  __shared__ float    Sf[64][66];
  __shared__ _Float16 Ps[64][72];
  __shared__ float    red[64][4];

  const int head  = blockIdx.x & 7;
  const int fixed = (blockIdx.x >> 3) & 63;
  const int b     = blockIdx.x >> 9;

  const size_t stride_s  = axis ? (size_t)64 * C_DIM : (size_t)C_DIM;
  const size_t off_fixed = (size_t)fixed * (axis ? (size_t)C_DIM
                                                 : (size_t)64 * C_DIM);
  const size_t org = (size_t)b * 64 * 64 * C_DIM + (size_t)head * KD + off_fixed;

  const int t = threadIdx.x;
  {  // stage: 64 rows x 64 f16 each for Q/K (row-major) and V (transposed)
    int    row = t >> 2, seg = (t & 3) * 16;
    size_t g   = org + (size_t)row * stride_s + seg;
    *(v8h*)&Qs[row][seg]     = *(const v8h*)(Qg + g);
    *(v8h*)&Qs[row][seg + 8] = *(const v8h*)(Qg + g + 8);
    *(v8h*)&Ks[row][seg]     = *(const v8h*)(Kg + g);
    *(v8h*)&Ks[row][seg + 8] = *(const v8h*)(Kg + g + 8);
    v8h v0 = *(const v8h*)(Vg + g);
    v8h v1 = *(const v8h*)(Vg + g + 8);
    #pragma unroll
    for (int j = 0; j < 8; ++j) {
      VsT[seg + j][row]     = v0[j];
      VsT[seg + 8 + j][row] = v1[j];
    }
  }
  __syncthreads();

  const int lane = t & 31, wave = t >> 5;
  const int hb = lane >> 4, l15 = lane & 15;

  // ---- S = Q K^T : 16 tiles of 16x16 over 8 waves (2 each)
  #pragma unroll
  for (int ti = 0; ti < 2; ++ti) {
    int tile = wave + ti * 8;
    int mi = tile >> 2, ni = tile & 3;
    v8f acc = {};
    #pragma unroll
    for (int kc = 0; kc < 64; kc += 32) {
      union { v16h v; v8h h[2]; } a, bf;
      const _Float16* qp = &Qs[mi * 16 + l15][kc + 8 * hb];
      a.h[0] = *(const v8h*)qp;
      a.h[1] = *(const v8h*)(qp + 16);
      const _Float16* kp = &Ks[ni * 16 + l15][kc + 8 * hb];  // B[d][j]=K[j][d]
      bf.h[0] = *(const v8h*)kp;
      bf.h[1] = *(const v8h*)(kp + 16);
      acc = wmma_f16(a.v, bf.v, acc);
    }
    #pragma unroll
    for (int r = 0; r < 8; ++r)
      Sf[mi * 16 + r + 8 * hb][ni * 16 + l15] = acc[r];
  }
  __syncthreads();

  // ---- softmax(+mask): 4 threads per row, 16 cols each
  {
    int row = t >> 2, c0 = (t & 3) * 16;
    const float* mrow = mask + ((size_t)head * 64 + row) * 64 + c0;
    float mx = -1e30f;
    #pragma unroll
    for (int j = 0; j < 16; ++j) {
      float v = Sf[row][c0 + j] + mrow[j];
      Sf[row][c0 + j] = v;
      mx = fmaxf(mx, v);
    }
    red[row][t & 3] = mx;
    __syncthreads();
    float rmx = fmaxf(fmaxf(red[row][0], red[row][1]),
                      fmaxf(red[row][2], red[row][3]));
    __syncthreads();
    float s = 0.f;
    #pragma unroll
    for (int j = 0; j < 16; ++j) {
      float e = __expf(Sf[row][c0 + j] - rmx);
      Sf[row][c0 + j] = e;
      s += e;
    }
    red[row][t & 3] = s;
    __syncthreads();
    float inv = 1.f / (red[row][0] + red[row][1] + red[row][2] + red[row][3]);
    #pragma unroll
    for (int j = 0; j < 16; ++j)
      Ps[row][c0 + j] = (_Float16)(Sf[row][c0 + j] * inv);
  }
  __syncthreads();

  // ---- O = P V
  #pragma unroll
  for (int ti = 0; ti < 2; ++ti) {
    int tile = wave + ti * 8;
    int mi = tile >> 2, ni = tile & 3;
    v8f acc = {};
    #pragma unroll
    for (int kc = 0; kc < 64; kc += 32) {
      union { v16h v; v8h h[2]; } a, bf;
      const _Float16* pp = &Ps[mi * 16 + l15][kc + 8 * hb];
      a.h[0] = *(const v8h*)pp;
      a.h[1] = *(const v8h*)(pp + 16);
      const _Float16* vp = &VsT[ni * 16 + l15][kc + 8 * hb];  // B[v][d]=VsT[d][v]
      bf.h[0] = *(const v8h*)vp;
      bf.h[1] = *(const v8h*)(vp + 16);
      acc = wmma_f16(a.v, bf.v, acc);
    }
    #pragma unroll
    for (int r = 0; r < 8; ++r) {
      int    row = mi * 16 + r + 8 * hb;   // query position
      int    col = ni * 16 + l15;          // d within head
      size_t g   = org + (size_t)row * stride_s + col;
      float  o   = acc[r];
      if (addend) o += addend[g];
      out16[g] = (_Float16)o;
    }
  }
}

// ---------------------------------------------------------------- launcher
extern "C" void kernel_launch(void* const* d_in, const int* in_sizes, int n_in,
                              void* d_out, int out_size, void* d_ws,
                              size_t ws_size, hipStream_t stream) {
  const float* x1     = (const float*)d_in[0];
  const float* x2     = (const float*)d_in[1];
  const float* sn     = (const float*)d_in[2];
  const float* cs     = (const float*)d_in[3];
  const float* mask_h = (const float*)d_in[4];
  const float* mask_w = (const float*)d_in[5];
  const float* Wq     = (const float*)d_in[6];
  const float* bq     = (const float*)d_in[7];
  const float* Wk     = (const float*)d_in[8];
  const float* bk     = (const float*)d_in[9];
  const float* Wv     = (const float*)d_in[10];
  const float* bv     = (const float*)d_in[11];
  const float* lw     = (const float*)d_in[12];
  const float* lb     = (const float*)d_in[13];
  const float* Wo     = (const float*)d_in[14];
  const float* bo     = (const float*)d_in[15];

  const size_t M  = (size_t)16 * 64 * 64;   // 65536 tokens
  const size_t MC = M * C_DIM;              // 33,554,432 elements
  const size_t WC = (size_t)C_DIM * C_DIM;

  char* wp = (char*)d_ws;
  auto alloc = [&](size_t bytes) -> char* {
    char* p = wp;
    wp += (bytes + 255) & ~(size_t)255;
    return p;
  };
  _Float16* x1h    = (_Float16*)alloc(MC * 2);
  _Float16* x2h    = (_Float16*)alloc(MC * 2);
  _Float16* wqt    = (_Float16*)alloc(WC * 2);   // transposed [N][K]
  _Float16* wkt    = (_Float16*)alloc(WC * 2);
  _Float16* wvt    = (_Float16*)alloc(WC * 2);
  _Float16* wot    = (_Float16*)alloc(WC * 2);
  _Float16* qrh    = (_Float16*)alloc(MC * 2);   // rope(q), f16
  _Float16* krh    = (_Float16*)alloc(MC * 2);   // rope(k*scale), f16
  float*    v32    = (float*)alloc(MC * 4);
  _Float16* vh     = (_Float16*)alloc(MC * 2);
  float*    lepe32 = (float*)alloc(MC * 4);
  _Float16* vvh    = (_Float16*)alloc(MC * 2);   // W-attention output
  _Float16* sumh   = (_Float16*)alloc(MC * 2);   // H-attn + lepe, f16

  dim3 blk(256);

  // operand converts (weights transposed for b128 LDS fragment reads)
  cvt_f32_f16<<<dim3(MC / 256), blk, 0, stream>>>(x1, x1h, MC);
  cvt_f32_f16<<<dim3(MC / 256), blk, 0, stream>>>(x2, x2h, MC);
  cvt_transpose_f16<<<dim3(WC / 256), blk, 0, stream>>>(Wq, wqt);
  cvt_transpose_f16<<<dim3(WC / 256), blk, 0, stream>>>(Wk, wkt);
  cvt_transpose_f16<<<dim3(WC / 256), blk, 0, stream>>>(Wv, wvt);
  cvt_transpose_f16<<<dim3(WC / 256), blk, 0, stream>>>(Wo, wot);

  dim3 ggrid(C_DIM / 64, (unsigned)(M / 128));
  // Q/K projections with fused RoPE epilogue (f16 out, no f32 round-trip)
  gemm_wmma_f16<<<ggrid, blk, 0, stream>>>(x1h, wqt, bq, nullptr, qrh, sn, cs,
                                           C_DIM, C_DIM, 1.0f, 1);
  gemm_wmma_f16<<<ggrid, blk, 0, stream>>>(x2h, wkt, bk, nullptr, krh, sn, cs,
                                           C_DIM, C_DIM, SCALING, 1);
  // V projection: f32 (for conv) + f16 (for attention) in one pass
  gemm_wmma_f16<<<ggrid, blk, 0, stream>>>(x1h, wvt, bv, v32, vh, nullptr,
                                           nullptr, C_DIM, C_DIM, 1.0f, 2);

  dwconv5x5<<<dim3(MC / 256), blk, 0, stream>>>(v32, lw, lb, lepe32, MC);

  // W-axis attention, then H-axis attention with fused (+lepe) epilogue
  dim3 agrid(16 * 64 * NH);
  attn_axis<<<agrid, blk, 0, stream>>>(qrh, krh, vh, mask_w, vvh, nullptr, 0);
  attn_axis<<<agrid, blk, 0, stream>>>(qrh, krh, vvh, mask_h, sumh, lepe32, 1);

  // (attn_out + lepe) @ Wo + bo -> d_out (f32)
  gemm_wmma_f16<<<ggrid, blk, 0, stream>>>(sumh, wot, bo, (float*)d_out,
                                           nullptr, nullptr, nullptr,
                                           C_DIM, C_DIM, 1.0f, 0);
}
